// LSTMNet_29506425323930
// MI455X (gfx1250) — matrix operand order repaired and checked by reading, not compiled, not run on hardware
//
#include <hip/hip_runtime.h>
#include <hip/hip_bf16.h>

// ---------------------------------------------------------------------------
// 2-layer LSTM (B=2048, T=512, IN=H=64) + FC head for MI455X (gfx1250).
//   * prep kernel packs weights into WMMA-B lane order (f16) + fused biases.
//   * main kernel: 64 blocks x 64 threads (2 waves). Each wave owns a 16-row
//     batch tile for all 512 steps. Weights staged into LDS via TDM
//     (tensor_load_to_lds). x[t] tiles double-buffered into LDS via
//     global_load_async_to_lds_b128. Gates via chained
//     v_wmma_f32_16x16x32_f16; c-state in VGPRs, h-state in per-wave LDS.
// ---------------------------------------------------------------------------

typedef _Float16 v16h __attribute__((ext_vector_type(16)));
typedef _Float16 v8h  __attribute__((ext_vector_type(8)));
typedef float    v8f  __attribute__((ext_vector_type(8)));
typedef unsigned int u32x4 __attribute__((ext_vector_type(4)));
typedef int          i32x4 __attribute__((ext_vector_type(4)));
typedef int          i32x8 __attribute__((ext_vector_type(8)));

#define BATCH   2048
#define SEQT    512
#define HID     64

// LDS layout (bytes):
//   [0, 131072)      : W0,W1,W2,W3 packed f16 (4 x 16384 halfs)
//   [131072, 133120) : bias0[256] f32, bias1[256] f32
//   [133120, ...)    : per-wave {xf32 double buffer 8KB, h1 2KB, h2 2KB}
#define WM_HALFS   16384
#define BIAS_BYTE  131072
#define WAVE_BYTE  133120
#define WAVE_SZ    12288
#define NWAVES     2
#define SMEM_BYTES (WAVE_BYTE + NWAVES * WAVE_SZ)   // 157696

// ---------------------------------------------------------------------------
// Prep: pack weights into per-lane B-operand order + fuse biases.
// B tile (32K x 16N) for v_wmma_f32_16x16x32_f16, lane l:
//   l <  16 : N = nt*16 + l,      halfs h=0..15 -> K = kt*32 + (h<8 ? h : h+8)
//   l >= 16 : N = nt*16 + l - 16, halfs h=0..15 -> K = kt*32 + (h<8 ? 8+h : 16+h)
// packed[((kt*16 + nt)*32 + lane)*16 + h]
// ---------------------------------------------------------------------------
__global__ void lstm_prep(const float* __restrict__ w_ih0,
                          const float* __restrict__ w_hh0,
                          const float* __restrict__ w_ih1,
                          const float* __restrict__ w_hh1,
                          const float* __restrict__ b_ih0,
                          const float* __restrict__ b_hh0,
                          const float* __restrict__ b_ih1,
                          const float* __restrict__ b_hh1,
                          char* __restrict__ ws) {
  int p = blockIdx.x * blockDim.x + threadIdx.x;
  if (p < 4 * WM_HALFS) {
    int m     = p >> 14;
    int local = p & (WM_HALFS - 1);
    const float* W = (m == 0) ? w_ih0 : (m == 1) ? w_hh0 : (m == 2) ? w_ih1 : w_hh1;
    int h    = local & 15;
    int lane = (local >> 4) & 31;
    int tile = local >> 9;
    int nt   = tile & 15;
    int kt   = tile >> 4;
    int N, K;
    if (lane < 16) { N = nt * 16 + lane;      K = kt * 32 + (h < 8 ? h : h + 8);      }
    else           { N = nt * 16 + lane - 16; K = kt * 32 + (h < 8 ? 8 + h : 16 + h); }
    ((_Float16*)ws)[p] = (_Float16)W[N * 64 + K];
  } else if (p < 4 * WM_HALFS + 512) {
    int i = p - 4 * WM_HALFS;
    float* bs = (float*)(ws + BIAS_BYTE);
    if (i < 256) bs[i] = b_ih0[i]       + b_hh0[i];
    else         bs[i] = b_ih1[i - 256] + b_hh1[i - 256];
  }
}

// ---------------------------------------------------------------------------
// Device helpers
// ---------------------------------------------------------------------------
static __device__ __forceinline__ v8f wmma16(v16h a, v16h b, v8f c) {
  return __builtin_amdgcn_wmma_f32_16x16x32_f16(false, a, false, b,
                                                (short)0, c, false, false);
}

// A operand (16Mx32K f16) from row-major f16 LDS buffer [16 rows][64 halfs].
static __device__ __forceinline__ v16h ldA(const _Float16* buf, int kt, int lane) {
  int row  = lane & 15;
  int kofs = (lane < 16) ? 0 : 8;
  const _Float16* p = buf + row * 64 + kt * 32 + kofs;
  v8h lo = *(const v8h*)p;
  v8h hi = *(const v8h*)(p + 16);
  return __builtin_shufflevector(lo, hi, 0, 1, 2, 3, 4, 5, 6, 7,
                                 8, 9, 10, 11, 12, 13, 14, 15);
}

// A operand from row-major f32 LDS buffer [16 rows][64 floats] (x staging),
// converting f32 -> f16 in VALU (co-executes with the WMMA pipe).
static __device__ __forceinline__ v16h ldAx(const float* buf, int kt, int lane) {
  int row  = lane & 15;
  int kofs = (lane < 16) ? 0 : 8;
  const float* p = buf + row * 64 + kt * 32 + kofs;
  float4 a0 = *(const float4*)p;
  float4 a1 = *(const float4*)(p + 4);
  float4 b0 = *(const float4*)(p + 16);
  float4 b1 = *(const float4*)(p + 20);
  v16h r = {(_Float16)a0.x, (_Float16)a0.y, (_Float16)a0.z, (_Float16)a0.w,
            (_Float16)a1.x, (_Float16)a1.y, (_Float16)a1.z, (_Float16)a1.w,
            (_Float16)b0.x, (_Float16)b0.y, (_Float16)b0.z, (_Float16)b0.w,
            (_Float16)b1.x, (_Float16)b1.y, (_Float16)b1.z, (_Float16)b1.w};
  return r;
}

// B operand from packed weight LDS: 32 contiguous bytes per lane per tile.
static __device__ __forceinline__ v16h ldB(const _Float16* w, int kt, int nt, int lane) {
  const _Float16* p = w + (((kt << 4) + nt) * 32 + lane) * 16;
  v8h lo = *(const v8h*)p;
  v8h hi = *(const v8h*)(p + 8);
  return __builtin_shufflevector(lo, hi, 0, 1, 2, 3, 4, 5, 6, 7,
                                 8, 9, 10, 11, 12, 13, 14, 15);
}

// Store a 16x16 h-chunk (C/D layout) as f16 into row-major [16][64] LDS.
static __device__ __forceinline__ void stH(_Float16* buf, int c0, int lane, v8f h) {
  int col  = c0 + (lane & 15);
  int roff = (lane < 16) ? 0 : 8;
#pragma unroll
  for (int r = 0; r < 8; ++r)
    buf[(r + roff) * 64 + col] = (_Float16)h[r];
}

static __device__ __forceinline__ v8f sig8(v8f x) {
  v8f r;
#pragma unroll
  for (int e = 0; e < 8; ++e) r[e] = 1.0f / (1.0f + __expf(-x[e]));
  return r;
}
static __device__ __forceinline__ v8f tanh8(v8f x) {
  v8f r;
#pragma unroll
  for (int e = 0; e < 8; ++e) {
    float ex = __expf(2.0f * x[e]);
    r[e] = (ex - 1.0f) / (ex + 1.0f);
  }
  return r;
}

// One LSTM layer step for one wave's 16-row tile. biasr[16] lives in VGPRs.
static __device__ __forceinline__ void lstm_layer_step(
    const _Float16* __restrict__ Wih, const _Float16* __restrict__ Whh,
    const float* __restrict__ biasr,
    v16h Ax0, v16h Ax1, v16h Ah0, v16h Ah1,
    v8f cst[4], _Float16* __restrict__ hbuf, int lane) {
#pragma unroll
  for (int c = 0; c < 4; ++c) {
    v8f g4[4];
#pragma unroll
    for (int g = 0; g < 4; ++g) {
      int nt = g * 4 + c;
      float bv = biasr[nt];
      v8f acc = {bv, bv, bv, bv, bv, bv, bv, bv};
      acc = wmma16(Ax0, ldB(Wih, 0, nt, lane), acc);
      acc = wmma16(Ax1, ldB(Wih, 1, nt, lane), acc);
      acc = wmma16(Ah0, ldB(Whh, 0, nt, lane), acc);
      acc = wmma16(Ah1, ldB(Whh, 1, nt, lane), acc);
      g4[g] = acc;
    }
    v8f ig = sig8(g4[0]);
    v8f fg = sig8(g4[1]);
    v8f gg = tanh8(g4[2]);
    v8f og = sig8(g4[3]);
    cst[c] = fg * cst[c] + ig * gg;
    v8f hv = og * tanh8(cst[c]);
    stH(hbuf, c * 16, lane, hv);
  }
}

// ---------------------------------------------------------------------------
// Main persistent kernel: 64 blocks x 64 threads (2 waves), wave per 16 rows.
// ---------------------------------------------------------------------------
__global__ void __launch_bounds__(64)
lstm_main(const float* __restrict__ x,
          const float* __restrict__ fc_w,
          const float* __restrict__ fc_b,
          const char* __restrict__ ws,
          float* __restrict__ out) {
  extern __shared__ char smem[];
  _Float16* W0 = (_Float16*)smem;
  _Float16* W1 = W0 + WM_HALFS;
  _Float16* W2 = W1 + WM_HALFS;
  _Float16* W3 = W2 + WM_HALFS;
  const float* Bs0 = (const float*)(smem + BIAS_BYTE);
  const float* Bs1 = Bs0 + 256;

  int tid  = threadIdx.x;
  int lane = tid & 31;
  int wid  = tid >> 5;

  char* wbase = smem + WAVE_BYTE + wid * WAVE_SZ;
  float*    xf0   = (float*)(wbase);            // 4KB  x[t] buffer 0 (f32)
  float*    xf1   = (float*)(wbase + 4096);     // 4KB  x[t] buffer 1 (f32)
  _Float16* h1buf = (_Float16*)(wbase + 8192);  // 2KB
  _Float16* h2buf = (_Float16*)(wbase + 10240); // 2KB

  // LDS byte offsets (flat-address low 32 bits == wave LDS offset).
  unsigned smemOff = (unsigned)(unsigned long long)(uintptr_t)smem;
  unsigned xfOff0  = smemOff + (unsigned)(WAVE_BYTE + wid * WAVE_SZ);
  unsigned xfOff1  = xfOff0 + 4096u;

  // ---- stage weights + biases into LDS (TDM if available) ----
#if __has_builtin(__builtin_amdgcn_tensor_load_to_lds)
  if (wid == 0) {
    unsigned long long ga = (unsigned long long)(uintptr_t)ws;
    // D# group 0: count=1, lds_addr, global_addr[56:0], type=2
    u32x4 g0 = {1u,
                smemOff,
                (unsigned)(ga & 0xffffffffull),
                (unsigned)((ga >> 32) & 0x01ffffffull) | (2u << 30)};
    // D# group 1: data_size=4B; tensor_dim0=33280 dwords; tensor_dim1=1;
    // tile_dim0=33280; tile_dim1=1; stride0=33280.
    const unsigned ND = (unsigned)(WAVE_BYTE / 4);   // 33280 dwords
    i32x8 g1 = {(int)(2u << 16),
                (int)((ND & 0xffffu) << 16),
                (int)(((ND >> 16) & 0xffffu) | (1u << 16)),
                (int)(ND << 16),
                (int)1,
                (int)ND,
                0, 0};
    i32x4 z4 = {0, 0, 0, 0};
#if defined(__clang_major__) && (__clang_major__ >= 23)
    i32x8 z8 = {0, 0, 0, 0, 0, 0, 0, 0};
    __builtin_amdgcn_tensor_load_to_lds(g0, g1, z4, z4, z8, 0);
#else
    __builtin_amdgcn_tensor_load_to_lds(g0, g1, z4, z4, 0);
#endif
    __builtin_amdgcn_s_wait_tensorcnt((unsigned short)0);
  }
#else
  {
    const uint4* src = (const uint4*)ws;
    uint4* dst = (uint4*)smem;
    const int n16 = WAVE_BYTE / 16;
    for (int i = tid; i < n16; i += blockDim.x) dst[i] = src[i];
  }
#endif

  // Zero this wave's h1/h2 (contiguous 4KB).
  {
    uint4 z = {0u, 0u, 0u, 0u};
    uint4* hz = (uint4*)h1buf;
    for (int i = lane; i < 256; i += 32) hz[i] = z;
  }
  __syncthreads();

  int b0 = (blockIdx.x * (blockDim.x >> 5) + wid) * 16;

  // x tile: lane pair per row (rows 0..15; 32 floats per half-row = 128B).
  int row2 = lane >> 1;
  int colh = (lane & 1) * 32;
  const float* xrow   = x + ((size_t)(b0 + row2) * SEQT) * 64 + colh;
  unsigned     lofs0  = xfOff0 + (unsigned)((row2 * 64 + colh) * 4);
  unsigned     lofs1  = xfOff1 + (unsigned)((row2 * 64 + colh) * 4);

  // Preload per-tile bias scalars into VGPRs (loop-invariant).
  float bias0r[16], bias1r[16];
#pragma unroll
  for (int nt = 0; nt < 16; ++nt) {
    bias0r[nt] = Bs0[nt * 16 + (lane & 15)];
    bias1r[nt] = Bs1[nt * 16 + (lane & 15)];
  }

  v8f zero8 = {0, 0, 0, 0, 0, 0, 0, 0};
  v8f c1[4], c2[4];
#pragma unroll
  for (int c = 0; c < 4; ++c) { c1[c] = zero8; c2[c] = zero8; }

  // Kick off async load of x[0] into buffer 0 (8 x 16B per lane = 128B).
  {
    const float* gp = xrow;
#pragma unroll
    for (int j = 0; j < 8; ++j)
      asm volatile("global_load_async_to_lds_b128 %0, %1, off offset:%2"
                   :: "v"(lofs0), "v"(gp), "i"(j * 16) : "memory");
  }

  for (int t = 0; t < SEQT; ++t) {
    // Current x buffer ready?
    asm volatile("s_wait_asynccnt 0x0" ::: "memory");

    // Prefetch x[t+1] into the other buffer while this step computes.
    if (t + 1 < SEQT) {
      const float* gp = xrow + (size_t)(t + 1) * 64;
      unsigned lofs = ((t + 1) & 1) ? lofs1 : lofs0;
#pragma unroll
      for (int j = 0; j < 8; ++j)
        asm volatile("global_load_async_to_lds_b128 %0, %1, off offset:%2"
                     :: "v"(lofs), "v"(gp), "i"(j * 16) : "memory");
    }

    const float* xcur = (t & 1) ? xf1 : xf0;

    // ---- layer 1: gates = x@Wih0^T + h1@Whh0^T + b0 ----
    v16h xA0  = ldAx(xcur, 0, lane),  xA1  = ldAx(xcur, 1, lane);
    v16h h1A0 = ldA(h1buf, 0, lane),  h1A1 = ldA(h1buf, 1, lane);
    lstm_layer_step(W0, W1, bias0r, xA0, xA1, h1A0, h1A1, c1, h1buf, lane);

    asm volatile("s_wait_dscnt 0" ::: "memory");

    // ---- layer 2: gates = h1_new@Wih1^T + h2@Whh1^T + b1 ----
    v16h n1A0 = ldA(h1buf, 0, lane), n1A1 = ldA(h1buf, 1, lane);
    v16h h2A0 = ldA(h2buf, 0, lane), h2A1 = ldA(h2buf, 1, lane);
    lstm_layer_step(W2, W3, bias1r, n1A0, n1A1, h2A0, h2A1, c2, h2buf, lane);

    asm volatile("s_wait_dscnt 0" ::: "memory");
  }

  // ---- FC head: out[16x4] = h2_last @ fc_w^T + fc_b ----
  if (lane < 16) {
    int row = lane;
    float hv[HID];
#pragma unroll
    for (int k = 0; k < HID; ++k) hv[k] = (float)h2buf[row * 64 + k];
#pragma unroll
    for (int cls = 0; cls < 4; ++cls) {
      float s = fc_b[cls];
      for (int k = 0; k < HID; ++k) s += hv[k] * fc_w[cls * 64 + k];
      out[(size_t)(b0 + row) * 4 + cls] = s;
    }
  }
}

// ---------------------------------------------------------------------------
// Host launch
// ---------------------------------------------------------------------------
extern "C" void kernel_launch(void* const* d_in, const int* in_sizes, int n_in,
                              void* d_out, int out_size, void* d_ws, size_t ws_size,
                              hipStream_t stream) {
  const float* x     = (const float*)d_in[0];
  const float* w_ih0 = (const float*)d_in[1];
  const float* w_hh0 = (const float*)d_in[2];
  const float* b_ih0 = (const float*)d_in[3];
  const float* b_hh0 = (const float*)d_in[4];
  const float* w_ih1 = (const float*)d_in[5];
  const float* w_hh1 = (const float*)d_in[6];
  const float* b_ih1 = (const float*)d_in[7];
  const float* b_hh1 = (const float*)d_in[8];
  const float* fc_w  = (const float*)d_in[9];
  const float* fc_b  = (const float*)d_in[10];
  float* out = (float*)d_out;
  char* ws = (char*)d_ws;

  int prep_threads = 4 * WM_HALFS + 512;
  lstm_prep<<<(prep_threads + 255) / 256, 256, 0, stream>>>(
      w_ih0, w_hh0, w_ih1, w_hh1, b_ih0, b_hh0, b_ih1, b_hh1, ws);

  // 64 workgroups x 2 waves, each wave owns 16 batch rows.
  lstm_main<<<BATCH / 32, 64, SMEM_BYTES, stream>>>(x, fc_w, fc_b, ws, out);
}
